// HierarchicalAttention_25202868093229
// MI455X (gfx1250) — matrix operand order repaired and checked by reading, not compiled, not run on hardware
//
#include <hip/hip_runtime.h>

// HierarchicalAttention for MI455X (gfx1250, wave32, WMMA + async-to-LDS).
//
// Exact algebraic simplifications vs the reference:
//  * seq_len==1 => softmax==1 => attn==v; q/k projections dead; only the
//    v-slice (rows [2D,3D)) of in_w matters.
//  * r_w/i_w/s_w/c_w outputs are identically 1.0.
//  * c_att never feeds any output => entire combined-branch MHA skipped.
//
// Per branch: v = x@Wv^T+bv ; out = v@Wo^T+bo+x ; att = LN(out) ;
// h = att@P^T+pb -> interleaved into combined_hidden.
//
// Numerics: every fp32 operand is pre-split ONCE into bf16 planes
// (hi = bf16(x), lo = bf16(x-hi)); GEMMs compute hi*hi + hi*lo + lo*hi with
// v_wmma_f32_16x16x32_bf16 (fp32 accum) => ~2^-16 relative error.
// GEMM inner loop: double-buffered global_load_async_to_lds_b128 (ASYNCcnt)
// overlapping 24 WMMAs per wave per K-step (32x64 wave tile, reuse factor 4).

typedef __bf16 v4bf  __attribute__((ext_vector_type(4)));
typedef __bf16 v8bf  __attribute__((ext_vector_type(8)));
typedef __bf16 v16bf __attribute__((ext_vector_type(16)));
typedef float  v4f   __attribute__((ext_vector_type(4)));
typedef float  v8f   __attribute__((ext_vector_type(8)));

#define BM 128
#define BN 128
#define BK 32
// LDS buffer (bytes): Ah 8192 | Al 8192 | Bh 8192 | Bl 8192  = 32768 per buffer
#define BUFB 32768

__device__ __forceinline__ v16bf ld_frag(const __bf16* p0, const __bf16* p1) {
  v8bf a = *(const v8bf*)p0;   // ds_load_b128
  v8bf b = *(const v8bf*)p1;   // ds_load_b128
  return __builtin_shufflevector(a, b, 0,1,2,3,4,5,6,7,8,9,10,11,12,13,14,15);
}

// One 16-byte async DMA: LDS[lds_byte] <= MEM[sbase + gbyte]  (GVS mode)
__device__ __forceinline__ void async_cp16(unsigned lds_byte,
                                           const __bf16* sbase, unsigned gbyte) {
  asm volatile("global_load_async_to_lds_b128 %0, %1, %2"
               :: "v"(lds_byte), "v"(gbyte), "s"(sbase) : "memory");
}

__device__ __forceinline__ void split_bf16(float x, __bf16& h, __bf16& l) {
  h = (__bf16)x;               // RNE
  l = (__bf16)(x - (float)h);  // residual
}

// C = A @ W^T + bias (+resid). A given as bf16 hi/lo planes (M,K) row-major,
// W as bf16 hi/lo planes (N,K) row-major. Output either fp32 (Cf, stride ldc,
// optional fp32 resid with stride N) or bf16 hi/lo planes (Chi/Clo, stride N).
// M%128==0, N%128==0, K%32==0.
__global__ __launch_bounds__(256) void gemm_planes_kernel(
    const __bf16* __restrict__ Ahi, const __bf16* __restrict__ Alo,
    const __bf16* __restrict__ Whi, const __bf16* __restrict__ Wlo,
    const float*  __restrict__ bias,
    const float*  __restrict__ resid,
    float*        __restrict__ Cf,
    __bf16*       __restrict__ Chi, __bf16* __restrict__ Clo,
    int N, int K, int ldc)
{
  __shared__ __align__(16) __bf16 smem[2 * (BUFB / 2)];

  const int tid  = threadIdx.x;
  const int lane = tid & 31;
  const int wave = tid >> 5;
  const int wm   = wave >> 1;   // 0..3 -> 32-row strip
  const int wn   = wave & 1;    // 0..1 -> 64-col strip
  const int half = lane >> 4;
  const int lr   = lane & 15;

  const int m0 = blockIdx.y * BM;
  const int n0 = blockIdx.x * BN;

  const __bf16* Ag_h = Ahi + (size_t)m0 * K;
  const __bf16* Ag_l = Alo + (size_t)m0 * K;
  const __bf16* Wg_h = Whi + (size_t)n0 * K;
  const __bf16* Wg_l = Wlo + (size_t)n0 * K;

  const unsigned lds_base = (unsigned)(uintptr_t)&smem[0];

  // copy geometry: 16B chunks; rows 0..127 per tile, 4 chunks per 32-elem row.
  // thread covers rows (tid>>2) and (tid>>2)+64, chunk tid&3, for all 4 planes.
  const int      arow  = tid >> 2;                        // 0..63
  const int      ac    = tid & 3;
  const unsigned goff0 = (unsigned)arow * (unsigned)(2 * K) + ac * 16;
  const unsigned goff1 = goff0 + 64u * (unsigned)(2 * K);  // rows 64..127
  const unsigned lofs0 = (unsigned)arow * 64 + ac * 16;
  const unsigned lofs1 = lofs0 + 4096;

  auto issue = [&](unsigned lb, unsigned kb) {   // kb = K-offset in bytes
    unsigned g0 = goff0 + kb, g1 = goff1 + kb;
    async_cp16(lb + 0     + lofs0, Ag_h, g0);
    async_cp16(lb + 0     + lofs1, Ag_h, g1);
    async_cp16(lb + 8192  + lofs0, Ag_l, g0);
    async_cp16(lb + 8192  + lofs1, Ag_l, g1);
    async_cp16(lb + 16384 + lofs0, Wg_h, g0);
    async_cp16(lb + 16384 + lofs1, Wg_h, g1);
    async_cp16(lb + 24576 + lofs0, Wg_l, g0);
    async_cp16(lb + 24576 + lofs1, Wg_l, g1);
  };

  v8f acc[2][4] = {};
  const int nk = K / BK;

  issue(lds_base, 0);
  for (int s = 0; s < nk; ++s) {
    asm volatile("s_wait_asynccnt 0x0" ::: "memory");
    __syncthreads();                               // buf s&1 fully visible
    if (s + 1 < nk)
      issue(lds_base + (unsigned)((s + 1) & 1) * BUFB,
            (unsigned)(s + 1) * (BK * 2));         // DMA overlaps WMMAs

    const __bf16* Ah = &smem[(s & 1) * (BUFB / 2)];
    const __bf16* Al = Ah + 4096;
    const __bf16* Bh = Ah + 8192;
    const __bf16* Bl = Ah + 12288;

    v16bf ah[2], al[2], bh[4], bl[4];
#pragma unroll
    for (int i = 0; i < 2; ++i) {
      // A 16x32 frag: lanes<16 hold K 0..7 & 16..23; lanes>=16 hold K 8..15 & 24..31
      int mr = (wm * 32 + i * 16 + lr) * BK;
      ah[i] = ld_frag(&Ah[mr + half * 8], &Ah[mr + half * 8 + 16]);
      al[i] = ld_frag(&Al[mr + half * 8], &Al[mr + half * 8 + 16]);
    }
#pragma unroll
    for (int j = 0; j < 4; ++j) {
      // B 32x16 frag: lane=N; lanes<16 hold K 0..7 & 16..23; lanes>=16 hold K 8..15 & 24..31
      int nr = (wn * 64 + j * 16 + lr) * BK;
      bh[j] = ld_frag(&Bh[nr + half * 8], &Bh[nr + half * 8 + 16]);
      bl[j] = ld_frag(&Bl[nr + half * 8], &Bl[nr + half * 8 + 16]);
    }
#pragma unroll
    for (int i = 0; i < 2; ++i)
#pragma unroll
      for (int j = 0; j < 4; ++j) {
        acc[i][j] = __builtin_amdgcn_wmma_f32_16x16x32_bf16(
            false, ah[i], false, bh[j], (short)0, acc[i][j], false, false);
        acc[i][j] = __builtin_amdgcn_wmma_f32_16x16x32_bf16(
            false, ah[i], false, bl[j], (short)0, acc[i][j], false, false);
        acc[i][j] = __builtin_amdgcn_wmma_f32_16x16x32_bf16(
            false, al[i], false, bh[j], (short)0, acc[i][j], false, false);
      }
    // no trailing barrier: next iteration's wait+barrier protects buffer reuse
  }

  // epilogue (C/D layout: lanes<16 -> M=r,N=lr ; lanes>=16 -> M=8+r,N=lr)
  if (Cf) {
    if (resid) {
#pragma unroll
      for (int i = 0; i < 2; ++i) {
        int rbase = m0 + wm * 32 + i * 16 + half * 8;
#pragma unroll
        for (int j = 0; j < 4; ++j) {
          int col = n0 + wn * 64 + j * 16 + lr;
          float bv = bias[col];
#pragma unroll
          for (int r = 0; r < 8; ++r) {
            int row = rbase + r;
            Cf[(size_t)row * ldc + col] =
                acc[i][j][r] + bv + resid[(size_t)row * N + col];
          }
        }
      }
    } else {
#pragma unroll
      for (int i = 0; i < 2; ++i) {
        int rbase = m0 + wm * 32 + i * 16 + half * 8;
#pragma unroll
        for (int j = 0; j < 4; ++j) {
          int col = n0 + wn * 64 + j * 16 + lr;
          float bv = bias[col];
#pragma unroll
          for (int r = 0; r < 8; ++r) {
            int row = rbase + r;
            Cf[(size_t)row * ldc + col] = acc[i][j][r] + bv;
          }
        }
      }
    }
  } else {
#pragma unroll
    for (int i = 0; i < 2; ++i) {
      int rbase = m0 + wm * 32 + i * 16 + half * 8;
#pragma unroll
      for (int j = 0; j < 4; ++j) {
        int col = n0 + wn * 64 + j * 16 + lr;
        float bv = bias[col];
#pragma unroll
        for (int r = 0; r < 8; ++r) {
          int row = rbase + r;
          float val = acc[i][j][r] + bv;
          __bf16 h, l;
          split_bf16(val, h, l);
          size_t idx = (size_t)row * N + col;
          Chi[idx] = h;
          Clo[idx] = l;
        }
      }
    }
  }
}

// Elementwise fp32 -> (hi,lo) bf16 planes.
__global__ __launch_bounds__(256) void cvt_split_kernel(
    const float* __restrict__ src, __bf16* __restrict__ hi,
    __bf16* __restrict__ lo, int n4)
{
  int i = blockIdx.x * 256 + threadIdx.x;
  if (i >= n4) return;
  v4f v = *(const v4f*)(src + (size_t)i * 4);
  v4bf h, l;
#pragma unroll
  for (int k = 0; k < 4; ++k) { __bf16 hh, ll; split_bf16(v[k], hh, ll); h[k] = hh; l[k] = ll; }
  *(v4bf*)(hi + (size_t)i * 4) = h;
  *(v4bf*)(lo + (size_t)i * 4) = l;
}

// LayerNorm over rows of 1024 fp32, emitting bf16 hi/lo planes directly.
__global__ __launch_bounds__(256) void ln1024_kernel(
    const float* __restrict__ x, __bf16* __restrict__ hi, __bf16* __restrict__ lo,
    const float* __restrict__ g, const float* __restrict__ b)
{
  __shared__ float rs[256], rss[256];
  const int tid = threadIdx.x;
  const float* p = x + (size_t)blockIdx.x * 1024;
  float v[4];
  float s = 0.f, ss = 0.f;
#pragma unroll
  for (int i = 0; i < 4; ++i) {
    float t = p[tid + i * 256];
    v[i] = t; s += t; ss += t * t;
  }
  rs[tid] = s; rss[tid] = ss;
  __syncthreads();
  for (int off = 128; off > 0; off >>= 1) {
    if (tid < off) { rs[tid] += rs[tid + off]; rss[tid] += rss[tid + off]; }
    __syncthreads();
  }
  float mean = rs[0] * (1.f / 1024.f);
  float var  = rss[0] * (1.f / 1024.f) - mean * mean;
  float inv  = rsqrtf(var + 1e-5f);
  size_t rowb = (size_t)blockIdx.x * 1024;
#pragma unroll
  for (int i = 0; i < 4; ++i) {
    int idx = tid + i * 256;
    float val = (v[i] - mean) * inv * g[idx] + b[idx];
    __bf16 h, l;
    split_bf16(val, h, l);
    hi[rowb + idx] = h;
    lo[rowb + idx] = l;
  }
}

__global__ void fill_ones_kernel(float* __restrict__ p, int n) {
  int i = blockIdx.x * blockDim.x + threadIdx.x;
  if (i < n) p[i] = 1.0f;
}

extern "C" void kernel_launch(void* const* d_in, const int* in_sizes, int n_in,
                              void* d_out, int out_size, void* d_ws, size_t ws_size,
                              hipStream_t stream) {
  (void)in_sizes; (void)n_in; (void)out_size;
  const int Bsz = 16384, D = 1024, HID = 512;
  const int LDO = 3 * HID;  // 1536

  const float* x[3] = { (const float*)d_in[0], (const float*)d_in[1], (const float*)d_in[2] };
  float* out = (float*)d_out;

  // ---- workspace layout ----
  // weights (per branch, overwritten): wv hi/lo (2MB each), wo hi/lo, p hi/lo (1MB each)
  char* wsb = (char*)d_ws;
  __bf16* wvh = (__bf16*)(wsb + 0);
  __bf16* wvl = (__bf16*)(wsb + (size_t)2  * 1048576);
  __bf16* woh = (__bf16*)(wsb + (size_t)4  * 1048576);
  __bf16* wol = (__bf16*)(wsb + (size_t)6  * 1048576);
  __bf16* pwh = (__bf16*)(wsb + (size_t)8  * 1048576);
  __bf16* pwl = (__bf16*)(wsb + (size_t)9  * 1048576);
  const size_t WEIGHT_BYTES = (size_t)10 * 1048576;

  // per-row chunk region: xh+xl (4KB) + vh+vl (4KB, reused as att planes) + out f32 (4KB)
  long long avail = (long long)ws_size - (long long)WEIGHT_BYTES;
  if (avail < 0) avail = 0;
  long long mr = avail / 12288;
  int R = (int)((mr / BM) * BM);
  if (R > Bsz) R = Bsz;
  if (R < BM)  R = BM;  // minimum viable chunk

  char* cb = wsb + WEIGHT_BYTES;
  __bf16* xh  = (__bf16*)cb;
  __bf16* xl  = xh + (size_t)R * D;
  __bf16* vh  = xl + (size_t)R * D;   // also att planes after LN
  __bf16* vl  = vh + (size_t)R * D;
  float*  obf = (float*)(vl + (size_t)R * D);

  dim3 blk(256);
  for (int p = 0; p < 3; ++p) {
    const float* in_w  = (const float*)d_in[3 + 4 * p];
    const float* in_b  = (const float*)d_in[4 + 4 * p];
    const float* out_w = (const float*)d_in[5 + 4 * p];
    const float* out_b = (const float*)d_in[6 + 4 * p];
    const float* ln_g  = (const float*)d_in[19 + 2 * p];
    const float* ln_b  = (const float*)d_in[20 + 2 * p];
    const float* pw    = (const float*)d_in[27 + 2 * p];
    const float* pb    = (const float*)d_in[28 + 2 * p];
    const float* Wv = in_w + (size_t)2 * D * D;  // v-slice of qkv weight
    const float* bv = in_b + 2 * D;

    // split weights once per branch
    int n4w = (D * D) / 4, n4p = (HID * D) / 4;
    cvt_split_kernel<<<dim3((n4w + 255) / 256), blk, 0, stream>>>(Wv,    wvh, wvl, n4w);
    cvt_split_kernel<<<dim3((n4w + 255) / 256), blk, 0, stream>>>(out_w, woh, wol, n4w);
    cvt_split_kernel<<<dim3((n4p + 255) / 256), blk, 0, stream>>>(pw,    pwh, pwl, n4p);

    for (int r0 = 0; r0 < Bsz; r0 += R) {
      int mc = Bsz - r0; if (mc > R) mc = R;
      const float* xc = x[p] + (size_t)r0 * D;

      int n4x = (mc * D) / 4;
      cvt_split_kernel<<<dim3((n4x + 255) / 256), blk, 0, stream>>>(xc, xh, xl, n4x);

      // v = x @ Wv^T + bv   -> bf16 planes
      gemm_planes_kernel<<<dim3(D / BN, mc / BM), blk, 0, stream>>>(
          xh, xl, wvh, wvl, bv, nullptr, nullptr, vh, vl, D, D, D);
      // out = v @ Wo^T + bo + x   -> fp32
      gemm_planes_kernel<<<dim3(D / BN, mc / BM), blk, 0, stream>>>(
          vh, vl, woh, wol, out_b, xc, obf, nullptr, nullptr, D, D, D);
      // att = LN(out)  -> bf16 planes (reuse v plane buffers; v is dead)
      ln1024_kernel<<<dim3(mc), blk, 0, stream>>>(obf, vh, vl, ln_g, ln_b);
      // h = att @ P^T + pb -> combined_hidden[:, p*HID:(p+1)*HID]
      gemm_planes_kernel<<<dim3(HID / BN, mc / BM), blk, 0, stream>>>(
          vh, vl, pwh, pwl, pb, nullptr,
          out + (size_t)r0 * LDO + (size_t)p * HID, nullptr, nullptr, HID, D, LDO);
    }
  }

  // r_w, i_w, s_w, c_w == 1.0 exactly (softmax over a single key)
  int ones_n = 4 * Bsz;
  fill_ones_kernel<<<dim3((ones_n + 255) / 256), blk, 0, stream>>>(
      out + (size_t)Bsz * LDO, ones_n);
}
